// DenseCrossBatchInfoNCE_89575837925852
// MI455X (gfx1250) — matrix operand order repaired and checked by reading, not compiled, not run on hardware
//
#include <hip/hip_runtime.h>
#include <hip/hip_bf16.h>

// ---------------------------------------------------------------------------
// Fused DenseCrossBatchInfoNCE for gfx1250 (MI455X).
//   B=16, T_MIX=256, T_REF=512, D=128  ->  N=4096 anchors, M=8192 refs.
// Pass 1: L2-normalize anchors/refs -> f16 workspace, per-anchor meta.
// Pass 2: tiled WMMA (v_wmma_f32_16x16x32_f16) GEMM fused with shifted
//         sum-exp (fixed shift 20: |cos|/tau + margin/tau < 18.2 for tau=0.07)
//         and block-local Gaussian soft-label KL terms. No logits in HBM.
// The per-batch "block" tiles (margin + Gaussian labels) form a contiguous
// 32-tile run inside exactly one wave's 128-tile range, so the tile loop is
// split into three branch-free segments (plain / block / plain).
// ---------------------------------------------------------------------------

typedef _Float16 v16h __attribute__((ext_vector_type(16)));
typedef _Float16 v8h  __attribute__((ext_vector_type(8)));
typedef float    v8f  __attribute__((ext_vector_type(8)));

#define NB 16
#define TMIX 256
#define TREF 512
#define DD 128
#define NANCH (NB * TMIX)   /* 4096 */
#define NREF  (NB * TREF)   /* 8192 */

// ---------------- Kernel 1: normalize + per-anchor meta ----------------
__global__ __launch_bounds__(128)
void prep_kernel(const float* __restrict__ m, const float* __restrict__ r,
                 const float* __restrict__ gt_coords,
                 const float* __restrict__ w_mix, const float* __restrict__ w_ref,
                 _Float16* __restrict__ Af, _Float16* __restrict__ Rf,
                 float* __restrict__ exf, int* __restrict__ rid,
                 float* __restrict__ aw) {
  int row = blockIdx.x;            // 0..4095 anchors, 4096..12287 refs
  int tid = threadIdx.x;           // 0..127 == D index
  bool is_anchor = row < NANCH;
  const float* src = is_anchor ? (m + (size_t)row * DD)
                               : (r + (size_t)(row - NANCH) * DD);
  float x = src[tid];
  float ss = x * x;
  #pragma unroll
  for (int off = 16; off > 0; off >>= 1) ss += __shfl_xor(ss, off, 32);
  __shared__ float red[4];
  int wv = tid >> 5, ln = tid & 31;
  if (ln == 0) red[wv] = ss;
  __syncthreads();
  float tot = red[0] + red[1] + red[2] + red[3];
  float nrm = fmaxf(sqrtf(tot), 1e-8f);
  _Float16 v = (_Float16)(x / nrm);
  if (is_anchor) Af[(size_t)row * DD + tid] = v;
  else           Rf[(size_t)(row - NANCH) * DD + tid] = v;

  if (is_anchor && tid == 0) {
    float exact = gt_coords[row] * ((float)TREF / 15.0f);   // / sec_per_frame
    int ri = (int)rintf(exact);                              // round-half-even
    ri = ri < 0 ? 0 : (ri > TREF - 1 ? TREF - 1 : ri);
    int b = row >> 8;                                        // row / T_MIX
    exf[row] = exact;
    rid[row] = ri;
    aw[row]  = w_mix[row] * w_ref[b * TREF + ri];
  }
}

// One 16x16 sim tile: A(16x128) x R_tile(128x16)^T via 4 K=32 WMMA steps.
static __device__ __forceinline__
v8f tile_dot(const v16h a[4], const _Float16* __restrict__ rbase, int koff_b) {
  v8f c = {};
  #pragma unroll
  for (int k = 0; k < 4; ++k) {
    v16h bc = *(const v16h*)(rbase + k * 32 + koff_b);
    c = __builtin_amdgcn_wmma_f32_16x16x32_f16(
        false, a[k], false, bc, (short)0, c, false, false);
  }
  return c;
}

// ---------------- Kernel 2: fused WMMA GEMM + softmax-KL ----------------
__global__ __launch_bounds__(128)
void infonce_kernel(const _Float16* __restrict__ Af, const _Float16* __restrict__ Rf,
                    const float* __restrict__ exf, const int* __restrict__ rid,
                    const float* __restrict__ aw,
                    const unsigned char* __restrict__ is_loop,
                    const float* __restrict__ chunk_frames,
                    const float* __restrict__ log_tau,
                    float* __restrict__ out) {
  const int tile     = blockIdx.x;      // 0..255 (16-anchor tiles)
  const int b        = tile >> 4;       // batch of this anchor tile
  const int row_base = tile << 4;       // global anchor index base
  const int tid  = threadIdx.x;
  const int wave = tid >> 5;
  const int lane = tid & 31;
  const int half = lane >> 4;           // 0: rows 0-7 / K-lo, 1: rows 8-15 / K-hi
  const int ln15 = lane & 15;

  float tau = __expf(log_tau[0]);
  tau = fminf(fmaxf(tau, 0.01f), 0.5f);
  const float inv_tau = 1.0f / tau;
  const float margin  = 0.25f * inv_tau;
  const bool  loopb   = is_loop[b] != 0;
  const float cf      = fmaxf(chunk_frames[b], 1.0f);
  const float rcf     = 1.0f / cf;

  // ---- load A tile (16 rows x 128 K) into WMMA A-operand registers ----
  const int arow = row_base + ln15;
  const int koff_a = half * 16;         // lanes 0-15: K 0-15, lanes 16-31: K 16-31
  const _Float16* abase = Af + (size_t)arow * DD;
  v16h a[4];
  #pragma unroll
  for (int c = 0; c < 4; ++c) {
    v8h lo = *(const v8h*)(abase + c * 32 + koff_a);
    v8h hi = *(const v8h*)(abase + c * 32 + koff_a + 8);
    a[c] = __builtin_shufflevector(lo, hi, 0,1,2,3,4,5,6,7,8,9,10,11,12,13,14,15);
  }

  // ---- per-row meta for this lane's 8 rows ----
  float exr[8], ridr[8];
  #pragma unroll
  for (int i = 0; i < 8; ++i) {
    int rr = row_base + i + half * 8;
    exr[i]  = exf[rr];
    ridr[i] = (float)rid[rr];
  }

  float S[8], Gs[8], GL[8], GSim[8];
  #pragma unroll
  for (int i = 0; i < 8; ++i) { S[i] = 0.f; Gs[i] = 0.f; GL[i] = 0.f; GSim[i] = 0.f; }

  const int koff_b = half * 16;         // lanes 0-15: K 0-15 of chunk, 16-31: K 16-31
  const int jt0 = wave * 128;           // each wave: 128 of 512 column tiles
  const int jtend = jt0 + 128;

  // The 32 same-batch tiles [32b, 32b+32) lie entirely inside one wave's range.
  const int  bs   = b << 5;
  const bool owns = (bs >= jt0) && (bs < jtend);
  const int  s0   = owns ? bs : jt0;    // block segment [s0, s1) (empty if !owns)
  const int  s1   = owns ? bs + 32 : jt0;

  const _Float16* rptr = Rf + ((size_t)jt0 * 16 + ln15) * DD;

  // ---- segment 1: plain tiles (branch-free hot loop) ----
  for (int jt = jt0; jt < s0; ++jt, rptr += 16 * DD) {
    v8f cacc = tile_dot(a, rptr, koff_b);
    #pragma unroll
    for (int i = 0; i < 8; ++i)
      S[i] += __expf(fmaf(cacc[i], inv_tau, -20.0f));
  }

  // ---- segment 2: same-batch block tiles (margin + Gaussian labels) ----
  for (int jt = s0; jt < s1; ++jt, rptr += 16 * DD) {
    v8f cacc = tile_dot(a, rptr, koff_b);
    const float t = (float)(((jt & 31) << 4) + ln15);        // col_t in [0,512)
    #pragma unroll
    for (int i = 0; i < 8; ++i) {
      float sim = cacc[i] * inv_tau;
      float lin = fabsf(ridr[i] - t);
      // cyclic distance without fmodf (exact: integer-valued operands < 512)
      float md  = fmaf(-floorf(lin * rcf), cf, lin);
      float cyc = fminf(md, cf - md);
      float td  = loopb ? cyc : lin;
      sim += ((td > 0.0f) && (td <= 3.0f)) ? margin : 0.0f;  // focal margin
      float d = t - exr[i];
      float e = -50.0f * d * d;                              // 1/(2*sigma^2), sigma=0.1
      float g = __expf(e);                                   // unnormalized soft label
      Gs[i]   += g;
      GL[i]   += g * e;                                      // sum g*log g
      GSim[i] += g * sim;
      S[i]    += __expf(sim - 20.0f);
    }
  }

  // ---- segment 3: remaining plain tiles ----
  for (int jt = s1; jt < jtend; ++jt, rptr += 16 * DD) {
    v8f cacc = tile_dot(a, rptr, koff_b);
    #pragma unroll
    for (int i = 0; i < 8; ++i)
      S[i] += __expf(fmaf(cacc[i], inv_tau, -20.0f));
  }

  // ---- reduce over the 16 lanes of each half (columns of the row) ----
  #pragma unroll
  for (int off = 1; off < 16; off <<= 1)
    #pragma unroll
    for (int i = 0; i < 8; ++i)
      S[i] += __shfl_xor(S[i], off, 32);

  if (owns) {   // wave-uniform: only the owning wave has nonzero label sums
    #pragma unroll
    for (int off = 1; off < 16; off <<= 1) {
      #pragma unroll
      for (int i = 0; i < 8; ++i) {
        Gs[i]   += __shfl_xor(Gs[i],   off, 32);
        GL[i]   += __shfl_xor(GL[i],   off, 32);
        GSim[i] += __shfl_xor(GSim[i], off, 32);
      }
    }
  }

  __shared__ float red[4][16][4];
  if (ln15 == 0) {
    #pragma unroll
    for (int i = 0; i < 8; ++i) {
      int row = i + half * 8;
      red[wave][row][0] = S[i];
      red[wave][row][1] = Gs[i];
      red[wave][row][2] = GL[i];
      red[wave][row][3] = GSim[i];
    }
  }
  __syncthreads();

  if (tid < 16) {
    float St = 0.f, Gst = 0.f, GLt = 0.f, GSt = 0.f;
    #pragma unroll
    for (int w = 0; w < 4; ++w) {
      St  += red[w][tid][0];
      Gst += red[w][tid][1];
      GLt += red[w][tid][2];
      GSt += red[w][tid][3];
    }
    float lse = __logf(St) + 20.0f;
    // KL = sum soft*log soft - sum soft*log_probs
    //    = GL/Gs - log(Gs) + lse - GSim/Gs        (sum soft == 1)
    float loss = GLt / Gst - __logf(Gst) + lse - GSt / Gst;
    loss *= aw[row_base + tid] * (1.0f / (float)NANCH);
    atomicAdd(out, loss);
  }
}

// ---------------------------------------------------------------------------
extern "C" void kernel_launch(void* const* d_in, const int* in_sizes, int n_in,
                              void* d_out, int out_size, void* d_ws, size_t ws_size,
                              hipStream_t stream) {
  (void)in_sizes; (void)n_in; (void)out_size; (void)ws_size;
  const float* m_centered   = (const float*)d_in[0];
  const float* r_centered   = (const float*)d_in[1];
  /* d_in[2] = gt_presence (all ones, unused) */
  const float* gt_coords    = (const float*)d_in[3];
  const float* w_mix        = (const float*)d_in[4];
  const float* w_ref        = (const float*)d_in[5];
  const unsigned char* is_loop = (const unsigned char*)d_in[6];  // jnp.bool_ = 1 byte
  const float* chunk_frames = (const float*)d_in[7];
  const float* log_tau      = (const float*)d_in[8];

  // Workspace layout (~3.05 MB):
  char* ws = (char*)d_ws;
  _Float16* Af = (_Float16*)ws;                                  // 4096*128*2 = 1 MiB
  _Float16* Rf = (_Float16*)(ws + (1u << 20));                   // 8192*128*2 = 2 MiB
  float*    exf = (float*)(ws + 3u * (1u << 20));                // 16 KiB
  int*      rid = (int*)  (ws + 3u * (1u << 20) + (16u << 10));  // 16 KiB
  float*    aw  = (float*)(ws + 3u * (1u << 20) + (32u << 10));  // 16 KiB

  hipMemsetAsync(d_out, 0, sizeof(float), stream);

  prep_kernel<<<NANCH + NREF, 128, 0, stream>>>(
      m_centered, r_centered, gt_coords, w_mix, w_ref, Af, Rf, exf, rid, aw);

  infonce_kernel<<<NANCH / 16, 128, 0, stream>>>(
      Af, Rf, exf, rid, aw, is_loop, chunk_frames, log_tau, (float*)d_out);
}
